// AttentionSACModel_8993661518716
// MI455X (gfx1250) — matrix operand order repaired and checked by reading, not compiled, not run on hardware
//
#include <hip/hip_runtime.h>
#include <hip/hip_bf16.h>
#include <math.h>

typedef __attribute__((ext_vector_type(16))) _Float16 v16h;
typedef __attribute__((ext_vector_type(8)))  float    v8f;

#define OWN_DIM   7
#define INTR_DIM  5
#define N_HEADS   3
#define HEAD_DIM  5
#define N_INTR    256
#define BATCH     32768
#define H1DIM     256
#define H2DIM     256
#define IN_DIM    22        // OWN_DIM + N_HEADS*HEAD_DIM
#define XPAD      32        // K padded for WMMA
#define OBS_COLS  (OWN_DIM + N_INTR * INTR_DIM)   // 1287
#define NEG_BIG   (-1e30f)
#define NEG_THR   (-1e29f)

// W1p: [nt(16)][lane(32)][e(16)] halves  -> 8192 halves (16 KB)
//   k = (e<8 ? (lane>>4)*8 + e : 16 + (lane>>4)*8 + e-8) (zero-padded past 22), n = nt*16 + (lane&15)
// W2p: [nt(16)][kk(8)][lane(32)][e(16)]  -> 65536 halves (128 KB)
//   k = kk*32 + (same in-slice k order as W1p),          n = nt*16 + (lane&15)
#define W1P_ELEMS (16 * 32 * 16)
#define W2P_ELEMS (16 * 8 * 32 * 16)

// CDNA5 has a hardware V_TANH_F32 transcendental (TRANS32 class, co-executes
// with VALU); use it when the toolchain exposes the builtin.
__device__ __forceinline__ float fast_tanh(float x) {
#if __has_builtin(__builtin_amdgcn_tanhf)
    return __builtin_amdgcn_tanhf(x);
#else
    return tanhf(x);
#endif
}

// ---------------------------------------------------------------------------
// Kernel 0: convert + pre-pack weights into per-lane WMMA B-fragment layout
// so each lane's B fragment is 32 contiguous bytes (2x global_load_b128).
// ---------------------------------------------------------------------------
__global__ __launch_bounds__(256) void pack_weights_kernel(
    const float* __restrict__ W1, const float* __restrict__ W2,
    _Float16* __restrict__ W1p, _Float16* __restrict__ W2p)
{
    int idx = blockIdx.x * 256 + threadIdx.x;
    if (idx < W1P_ELEMS) {
        int e  = idx & 15;
        int l  = (idx >> 4) & 31;
        int nt = idx >> 9;
        int k  = (e < 8) ? (l >> 4) * 8 + e : 16 + (l >> 4) * 8 + (e - 8);
        int n  = nt * 16 + (l & 15);
        W1p[idx] = (k < IN_DIM) ? (_Float16)W1[k * H1DIM + n] : (_Float16)0.f;
    } else {
        int j = idx - W1P_ELEMS;
        if (j < W2P_ELEMS) {
            int e  = j & 15;
            int l  = (j >> 4) & 31;
            int kk = (j >> 9) & 7;
            int nt = j >> 12;
            int k  = kk * 32 + ((e < 8) ? (l >> 4) * 8 + e : 16 + (l >> 4) * 8 + (e - 8));
            int n  = nt * 16 + (l & 15);
            W2p[j] = (_Float16)W2[k * H2DIM + n];
        }
    }
}

// ---------------------------------------------------------------------------
// Kernel 1: attention (online softmax) + LayerNorm -> X f16 (BATCH x 32)
// One wave32 per batch row; each lane owns 8 interactions.
// ---------------------------------------------------------------------------
__global__ __launch_bounds__(256) void attn_ln_kernel(
    const float* __restrict__ obs,
    const float* __restrict__ Wq,  const float* __restrict__ bq,
    const float* __restrict__ Wk,  const float* __restrict__ bk,
    const float* __restrict__ Wv,  const float* __restrict__ bv,
    const float* __restrict__ v_att, const float* __restrict__ temperature,
    const float* __restrict__ ln_g, const float* __restrict__ ln_b,
    _Float16* __restrict__ X)
{
    __shared__ float sWq[N_HEADS * OWN_DIM * HEAD_DIM];   // 105
    __shared__ float sbq[N_HEADS * HEAD_DIM];
    __shared__ float sWk[N_HEADS * INTR_DIM * HEAD_DIM];  // 75
    __shared__ float sbk[N_HEADS * HEAD_DIM];
    __shared__ float sWv[N_HEADS * INTR_DIM * HEAD_DIM];  // 75
    __shared__ float sbv[N_HEADS * HEAD_DIM];
    __shared__ float sva[N_HEADS * HEAD_DIM];
    __shared__ float sg[IN_DIM], sb[IN_DIM];

    int tid = threadIdx.x;
    for (int i = tid; i < 105; i += 256) sWq[i] = Wq[i];
    for (int i = tid; i < 75;  i += 256) { sWk[i] = Wk[i]; sWv[i] = Wv[i]; }
    for (int i = tid; i < 15;  i += 256) {
        sbq[i] = bq[i]; sbk[i] = bk[i]; sbv[i] = bv[i]; sva[i] = v_att[i];
    }
    for (int i = tid; i < IN_DIM; i += 256) { sg[i] = ln_g[i]; sb[i] = ln_b[i]; }
    __syncthreads();

    int lane = tid & 31;
    int row  = blockIdx.x * 8 + (tid >> 5);
    const float* orow = obs + (long long)row * OBS_COLS;

    float own[OWN_DIM];
#pragma unroll
    for (int i = 0; i < OWN_DIM; i++) own[i] = orow[i];

    // q = own @ Wq + bq  (every lane keeps the full 15-vector)
    float q[N_HEADS * HEAD_DIM];
#pragma unroll
    for (int h = 0; h < N_HEADS; h++)
#pragma unroll
        for (int d = 0; d < HEAD_DIM; d++) {
            float a = sbq[h * HEAD_DIM + d];
#pragma unroll
            for (int o = 0; o < OWN_DIM; o++)
                a = fmaf(own[o], sWq[(h * OWN_DIM + o) * HEAD_DIM + d], a);
            q[h * HEAD_DIM + d] = a;
        }

    const float sc = 0.4472135954999579f * fabsf(temperature[0]); // 1/sqrt(5)*|T|

    float m[N_HEADS], ssum[N_HEADS], ctx[N_HEADS * HEAD_DIM];
#pragma unroll
    for (int h = 0; h < N_HEADS; h++) { m[h] = NEG_BIG; ssum[h] = 0.f; }
#pragma unroll
    for (int i = 0; i < N_HEADS * HEAD_DIM; i++) ctx[i] = 0.f;

    for (int it = 0; it < N_INTR / 32; ++it) {
        int n = it * 32 + lane;
        const float* e = orow + OWN_DIM + n * INTR_DIM;
        float x0 = e[0], x1 = e[1], x2 = e[2], x3 = e[3], x4 = e[4];
        float ab = fabsf(x0) + fabsf(x1) + fabsf(x2) + fabsf(x3) + fabsf(x4);
        if (ab >= 1e-6f) {
#pragma unroll
            for (int h = 0; h < N_HEADS; h++) {
                float vd[HEAD_DIM], s = 0.f;
#pragma unroll
                for (int d = 0; d < HEAD_DIM; d++) {
                    const float* wk = &sWk[h * 25 + d];
                    const float* wv = &sWv[h * 25 + d];
                    float kd = sbk[h * 5 + d];
                    kd = fmaf(x0, wk[0],  kd); kd = fmaf(x1, wk[5],  kd);
                    kd = fmaf(x2, wk[10], kd); kd = fmaf(x3, wk[15], kd);
                    kd = fmaf(x4, wk[20], kd);
                    float vv = sbv[h * 5 + d];
                    vv = fmaf(x0, wv[0],  vv); vv = fmaf(x1, wv[5],  vv);
                    vv = fmaf(x2, wv[10], vv); vv = fmaf(x3, wv[15], vv);
                    vv = fmaf(x4, wv[20], vv);
                    vd[d] = vv;
                    s = fmaf(sva[h * 5 + d], fast_tanh(q[h * 5 + d] + kd), s);
                }
                s *= sc;
                // online softmax update
                float mn = fmaxf(m[h], s);
                float c  = __expf(m[h] - mn);   // underflows to 0 on first hit
                float p  = __expf(s - mn);
                ssum[h] = ssum[h] * c + p;
#pragma unroll
                for (int d = 0; d < HEAD_DIM; d++)
                    ctx[h * 5 + d] = ctx[h * 5 + d] * c + p * vd[d];
                m[h] = mn;
            }
        }
    }

    // cross-lane merge of (m, sum, ctx) per head — butterfly, all lanes converge
#pragma unroll
    for (int h = 0; h < N_HEADS; h++) {
#pragma unroll
        for (int off = 16; off > 0; off >>= 1) {
            float mo = __shfl_xor(m[h], off, 32);
            float so = __shfl_xor(ssum[h], off, 32);
            float co[HEAD_DIM];
#pragma unroll
            for (int d = 0; d < HEAD_DIM; d++)
                co[d] = __shfl_xor(ctx[h * 5 + d], off, 32);
            float mn = fmaxf(m[h], mo);
            float c1 = (m[h] > NEG_THR) ? __expf(m[h] - mn) : 0.f;
            float c2 = (mo   > NEG_THR) ? __expf(mo   - mn) : 0.f;
            ssum[h] = ssum[h] * c1 + so * c2;
#pragma unroll
            for (int d = 0; d < HEAD_DIM; d++)
                ctx[h * 5 + d] = ctx[h * 5 + d] * c1 + co[d] * c2;
            m[h] = mn;
        }
        float inv = (ssum[h] > 0.f) ? 1.f / ssum[h] : 0.f; // all-pad row -> 0
#pragma unroll
        for (int d = 0; d < HEAD_DIM; d++) ctx[h * 5 + d] *= inv;
    }

    // LayerNorm over [own, ctx] (22 elems), each lane emits element == lane
    float xv[IN_DIM];
#pragma unroll
    for (int i = 0; i < OWN_DIM; i++) xv[i] = own[i];
#pragma unroll
    for (int i = 0; i < N_HEADS * HEAD_DIM; i++) xv[OWN_DIM + i] = ctx[i];
    float mu = 0.f;
#pragma unroll
    for (int i = 0; i < IN_DIM; i++) mu += xv[i];
    mu *= (1.f / IN_DIM);
    float var = 0.f;
#pragma unroll
    for (int i = 0; i < IN_DIM; i++) { float d = xv[i] - mu; var = fmaf(d, d, var); }
    var *= (1.f / IN_DIM);
    float rstd = rsqrtf(var + 1e-5f);

    _Float16 outh = (_Float16)0.f;
    if (lane < IN_DIM)
        outh = (_Float16)((xv[lane] - mu) * rstd * sg[lane] + sb[lane]);
    X[(long long)row * XPAD + lane] = outh;
}

// ---------------------------------------------------------------------------
// Kernel 2: GEMM1  X(32768x32) @ W1 + b1, leaky(0.2) -> H1 f16
// One wave per 16-row tile; A loaded once (2x b128), 16 N-tiles, 16 WMMAs.
// B fragment = one contiguous 32B per-lane load from packed W1p.
// ---------------------------------------------------------------------------
__global__ __launch_bounds__(256) void gemm1_kernel(
    const _Float16* __restrict__ X, const _Float16* __restrict__ W1p,
    const float* __restrict__ b1, _Float16* __restrict__ Hout)
{
    int wave  = blockIdx.x * 8 + (threadIdx.x >> 5);  // 2048 waves
    int lane  = threadIdx.x & 31;
    int mbase = wave * 16;
    int mrow  = lane & 15;
    int hi    = lane >> 4;       // lane half selects the +8/+24 K groups
    int n     = lane & 15;

    // A fragment per ISA 16-bit A layout: K pairs are contiguous dwords.
    union { v16h v; uint4 q[2]; } A;
    const unsigned int* Xu = (const unsigned int*)X;
    long long rowp = (long long)(mbase + mrow) * (XPAD / 2);
    A.q[0] = *(const uint4*)(Xu + rowp + hi * 4);
    A.q[1] = *(const uint4*)(Xu + rowp + 8 + hi * 4);

    const v16h* Bp = (const v16h*)W1p;   // [nt][lane] fragments
#pragma unroll
    for (int nt = 0; nt < 16; ++nt) {
        int nb = nt * 16 + n;
        v16h bfrag = Bp[nt * 32 + lane];
        v8f c = {};
        c = __builtin_amdgcn_wmma_f32_16x16x32_f16(
                false, A.v, false, bfrag, (short)0, c, false, false);
        float bias = b1[nb];
#pragma unroll
        for (int i = 0; i < 8; i++) {
            float y = c[i] + bias;
            y = y > 0.f ? y : 0.2f * y;
            Hout[(long long)(mbase + i + hi * 8) * H1DIM + nb] = (_Float16)y;
        }
    }
}

// ---------------------------------------------------------------------------
// Kernel 3: GEMM2  H1(32768x256) @ W2 + b2, leaky -> H2 f16
// Wave computes 16x64 tile: 8 K-steps x 4 N-tiles = 32 WMMAs, A reuse x4.
// Inner loop: 2x b128 (A) + 2x b128 (B) + 1 wmma.
// ---------------------------------------------------------------------------
__global__ __launch_bounds__(256) void gemm2_kernel(
    const _Float16* __restrict__ Hin, const _Float16* __restrict__ W2p,
    const float* __restrict__ b2, _Float16* __restrict__ Hout)
{
    int wave  = blockIdx.x * 8 + (threadIdx.x >> 5);  // 8192 waves
    int lane  = threadIdx.x & 31;
    int mtile = wave >> 2;
    int nquad = wave & 3;
    int mbase = mtile * 16;
    int mrow  = lane & 15;
    int hi    = lane >> 4;
    int n     = lane & 15;

    v8f acc[4] = {};
    const unsigned int* Hu = (const unsigned int*)Hin;
    const v16h* Bp = (const v16h*)W2p;   // [nt][kk][lane] fragments
    long long rowp = (long long)(mbase + mrow) * (H1DIM / 2);

#pragma unroll
    for (int kk = 0; kk < 8; ++kk) {
        if (kk < 7)  // prefetch next K-slice of A (global_prefetch_b8)
            __builtin_prefetch(Hin + (rowp + (kk + 1) * 16) * 2, 0, 3);
        union { v16h v; uint4 q[2]; } A;
        long long base = rowp + kk * 16;
        A.q[0] = *(const uint4*)(Hu + base + hi * 4);
        A.q[1] = *(const uint4*)(Hu + base + 8 + hi * 4);
#pragma unroll
        for (int j = 0; j < 4; j++) {
            int nt = nquad * 4 + j;
            v16h bfrag = Bp[(nt * 8 + kk) * 32 + lane];
            acc[j] = __builtin_amdgcn_wmma_f32_16x16x32_f16(
                         false, A.v, false, bfrag, (short)0, acc[j], false, false);
        }
    }
#pragma unroll
    for (int j = 0; j < 4; j++) {
        int nb = (nquad * 4 + j) * 16 + n;
        float bias = b2[nb];
#pragma unroll
        for (int i = 0; i < 8; i++) {
            float y = acc[j][i] + bias;
            y = y > 0.f ? y : 0.2f * y;
            Hout[(long long)(mbase + i + hi * 8) * H2DIM + nb] = (_Float16)y;
        }
    }
}

// ---------------------------------------------------------------------------
// Kernel 4: head  H2(32768x256) @ Wf(256x2) + bf ; append log_std
// One wave per row; wave reduction over K.
// ---------------------------------------------------------------------------
__global__ __launch_bounds__(256) void final_kernel(
    const _Float16* __restrict__ H2p, const float* __restrict__ Wf,
    const float* __restrict__ bfv, const float* __restrict__ log_std,
    float* __restrict__ out)
{
    int wave = blockIdx.x * 8 + (threadIdx.x >> 5);
    int lane = threadIdx.x & 31;
    float a0 = 0.f, a1 = 0.f;
#pragma unroll
    for (int it = 0; it < H2DIM / 32; ++it) {
        int k = it * 32 + lane;
        float h = (float)H2p[(long long)wave * H2DIM + k];
        a0 = fmaf(h, Wf[k * 2 + 0], a0);
        a1 = fmaf(h, Wf[k * 2 + 1], a1);
    }
#pragma unroll
    for (int off = 16; off > 0; off >>= 1) {
        a0 += __shfl_xor(a0, off, 32);
        a1 += __shfl_xor(a1, off, 32);
    }
    if (lane == 0) {
        out[(long long)wave * 4 + 0] = a0 + bfv[0];
        out[(long long)wave * 4 + 1] = a1 + bfv[1];
        out[(long long)wave * 4 + 2] = log_std[0];
        out[(long long)wave * 4 + 3] = log_std[1];
    }
}

// ---------------------------------------------------------------------------
extern "C" void kernel_launch(void* const* d_in, const int* in_sizes, int n_in,
                              void* d_out, int out_size, void* d_ws, size_t ws_size,
                              hipStream_t stream) {
    (void)in_sizes; (void)n_in; (void)out_size; (void)ws_size;
    const float* obs   = (const float*)d_in[0];
    const float* Wq    = (const float*)d_in[1];
    const float* bq    = (const float*)d_in[2];
    const float* Wk    = (const float*)d_in[3];
    const float* bk    = (const float*)d_in[4];
    const float* Wv    = (const float*)d_in[5];
    const float* bv    = (const float*)d_in[6];
    const float* v_att = (const float*)d_in[7];
    const float* temp  = (const float*)d_in[8];
    const float* ln_g  = (const float*)d_in[9];
    const float* ln_b  = (const float*)d_in[10];
    const float* W1    = (const float*)d_in[11];
    const float* b1    = (const float*)d_in[12];
    const float* W2    = (const float*)d_in[13];
    const float* b2    = (const float*)d_in[14];
    const float* Wf    = (const float*)d_in[15];
    const float* bfv   = (const float*)d_in[16];
    const float* lstd  = (const float*)d_in[17];
    float* out = (float*)d_out;

    char* ws = (char*)d_ws;
    _Float16* X   = (_Float16*)(ws);                              //  2 MB
    _Float16* Hb1 = (_Float16*)(ws + (2ull  << 20));              // 16 MB
    _Float16* Hb2 = (_Float16*)(ws + (18ull << 20));              // 16 MB
    _Float16* W1p = (_Float16*)(ws + (34ull << 20));              // 16 KB
    _Float16* W2p = (_Float16*)(ws + (34ull << 20) + (1ull << 16)); // 128 KB

    pack_weights_kernel<<<(W1P_ELEMS + W2P_ELEMS) / 256, 256, 0, stream>>>(
        W1, W2, W1p, W2p);
    attn_ln_kernel<<<BATCH / 8, 256, 0, stream>>>(
        obs, Wq, bq, Wk, bk, Wv, bv, v_att, temp, ln_g, ln_b, X);
    gemm1_kernel<<<(BATCH / 16) / 8, 256, 0, stream>>>(X, W1p, b1, Hb1);
    gemm2_kernel<<<(BATCH / 16) * 4 / 8, 256, 0, stream>>>(Hb1, W2p, b2, Hb2);
    final_kernel<<<BATCH / 8, 256, 0, stream>>>(Hb2, Wf, bfv, lstd, out);
}
